// SequenceLSTMDecoder_5033701671667
// MI455X (gfx1250) — compile-verified, matrix operand order checked by reading, not run on hardware
//
#include <hip/hip_runtime.h>
#include <hip/hip_bf16.h>
#include <math.h>
#include <stdint.h>

// ---------------------------------------------------------------------------
// SequenceLSTMDecoder for MI455X (gfx1250, wave32, WMMA).
// B=32, T=128, H=512, V=32000.  Dominant cost: tied projection h@E^T
// (134 of 151 GFLOP) -> large bf16 WMMA GEMM with async-LDS staged B and
// non-temporal fp32 output stores (keep E resident in the 192MB L2 while the
// 524MB logits stream to HBM at 23.3 TB/s).
// ---------------------------------------------------------------------------

typedef __bf16 bf16_t;
typedef __bf16 v16bf __attribute__((ext_vector_type(16)));
typedef __bf16 v8bf  __attribute__((ext_vector_type(8)));
typedef float  v8f   __attribute__((ext_vector_type(8)));

#define BB 32
#define TT 128
#define HH 512
#define VV 32000
#define G4H 2048       // 4*H
#define SPAD 40        // LDS row pitch (32 data + 8 pad) -> bank-conflict-free

// Fragment loader matching CDNA5 16-bit A/B WMMA layout (ISA 7.12.2):
// lanes 0-15 hold K = {0..7, 16..23}, lanes 16-31 hold K = {8..15, 24..31};
// p must already point at (row, kbase + khalf*8). Two 16-byte chunks.
__device__ __forceinline__ v16bf load_frag16(const bf16_t* p) {
  union { v16bf v; v8bf h[2]; } u;
  u.h[0] = *(const v8bf*)(p);
  u.h[1] = *(const v8bf*)(p + 16);
  return u.v;
}

__device__ __forceinline__ v8f wmma_bf16(v16bf a, v16bf b, v8f c) {
  return __builtin_amdgcn_wmma_f32_16x16x32_bf16(
      /*neg_a=*/false, a, /*neg_b=*/false, b,
      /*c_mod=*/(short)0, c, /*reuse_a=*/false, /*reuse_b=*/false);
}

__device__ __forceinline__ float sigmoidf_(float x) {
  return 1.0f / (1.0f + __expf(-x));
}

// CDNA5 async global->LDS copy (ASYNCcnt) — per-lane 16B, whole-wave 512B.
// lds_off = LDS byte offset (low 32 bits of a generic pointer to __shared__).
__device__ __forceinline__ void async_copy_b128(uint32_t lds_off,
                                                const void* gaddr) {
  asm volatile("global_load_async_to_lds_b128 %0, %1, off"
               :: "v"(lds_off), "v"(gaddr)
               : "memory");
}
__device__ __forceinline__ void wait_async_le2() {
  asm volatile("s_wait_asynccnt 0x2" ::: "memory");
}
__device__ __forceinline__ void wait_async_0() {
  asm volatile("s_wait_asynccnt 0x0" ::: "memory");
}
__device__ __forceinline__ uint32_t lds_offset_of(const void* p) {
  return (uint32_t)(uintptr_t)p;   // generic->LDS: low 32 bits hold ds offset
}

// --- K0: fp32 -> bf16 conversion (E, W_ih, W_hh) ---------------------------
__global__ __launch_bounds__(256) void k_cvt_bf16(const float* __restrict__ in,
                                                  bf16_t* __restrict__ out,
                                                  size_t n) {
  size_t i = (size_t)blockIdx.x * blockDim.x + threadIdx.x;
  size_t stride = (size_t)gridDim.x * blockDim.x;
  for (; i < n; i += stride) out[i] = (bf16_t)in[i];
}

// --- K1: embedding gather with teacher-forcing shift -----------------------
// row r = t*B + b ; token = (t==0) ? START_IDX(1) : target[b][t-1]
__global__ __launch_bounds__(256) void k_gather_x(const float* __restrict__ E,
                                                  const int* __restrict__ target,
                                                  bf16_t* __restrict__ X) {
  int r = blockIdx.x;                 // 0..4095
  int t = r >> 5, b = r & 31;
  int tok = (t == 0) ? 1 : target[b * TT + (t - 1)];
  const float* src = E + (size_t)tok * HH;
  bf16_t* dst = X + (size_t)r * HH;
  for (int k = threadIdx.x; k < HH; k += blockDim.x) dst[k] = (bf16_t)src[k];
}

// --- K2: Gx = X @ W_ih^T + (b_ih + b_hh), [4096 x 2048], K=512 -------------
// 256 threads = 8 waves; WG tile 128x128; wave w -> rows w*16..w*16+15.
// Gx is written non-temporally: streamed once, must not evict E from L2.
__global__ __launch_bounds__(256) void k_gemm_gates(
    const bf16_t* __restrict__ X, const bf16_t* __restrict__ Wih,
    const float* __restrict__ b_ih, const float* __restrict__ b_hh,
    float* __restrict__ Gx) {
  const int wave = threadIdx.x >> 5;
  const int lane = threadIdx.x & 31;
  const int l15 = lane & 15;
  const int khalf = lane >> 4;
  const int m0 = blockIdx.y * 128 + wave * 16;
  const int n0 = blockIdx.x * 128;

  v8f acc[8];
  for (int s = 0; s < 8; ++s) acc[s] = (v8f)(0.0f);

  const bf16_t* pa_base = X + (size_t)(m0 + l15) * HH + khalf * 8;
  for (int k = 0; k < HH; k += 32) {
    v16bf a = load_frag16(pa_base + k);
#pragma unroll
    for (int s = 0; s < 8; ++s) {
      const bf16_t* pb = Wih + (size_t)(n0 + s * 16 + l15) * HH + k + khalf * 8;
      acc[s] = wmma_bf16(a, load_frag16(pb), acc[s]);
    }
  }
  // D layout: VGPR v, lanes 0-15 -> row v, lanes 16-31 -> row v+8, col = l15.
#pragma unroll
  for (int s = 0; s < 8; ++s) {
    int col = n0 + s * 16 + l15;
    float bias = b_ih[col] + b_hh[col];
#pragma unroll
    for (int v = 0; v < 8; ++v) {
      int row = m0 + v + khalf * 8;
      __builtin_nontemporal_store(acc[s][v] + bias,
                                  &Gx[(size_t)row * G4H + col]);
    }
  }
}

// --- K3: sequential LSTM recurrence, single WGP, 32 waves ------------------
// Wave w owns hidden cols j0=w*16..j0+15 for ALL four gates -> i/f/g/o are
// in-register; c stays in registers; h (bf16, 32KB) in LDS for cross-wave
// reuse by the next step's h @ W_hh^T.
__global__ __launch_bounds__(1024) void k_recurrence(
    const float* __restrict__ h0, const float* __restrict__ Gx,
    const bf16_t* __restrict__ Whh, bf16_t* __restrict__ Hall) {
  __shared__ bf16_t h_lds[BB * HH];   // 32 KB

  const int tid = threadIdx.x;
  const int wave = tid >> 5;          // 0..31
  const int lane = tid & 31;
  const int l15 = lane & 15;
  const int khalf = lane >> 4;
  const int j0 = wave * 16;           // hidden column block

  // init h (LDS) and c (registers) from encoder state
  for (int i = tid; i < BB * HH; i += 1024) h_lds[i] = (bf16_t)h0[i];
  float creg[16];                      // [mt][v] : row = mt*16 + v + khalf*8
#pragma unroll
  for (int mt = 0; mt < 2; ++mt)
#pragma unroll
    for (int v = 0; v < 8; ++v) {
      int row = mt * 16 + v + khalf * 8;
      creg[mt * 8 + v] = h0[(size_t)row * HH + j0 + l15];
    }
  __syncthreads();

  for (int t = 0; t < TT; ++t) {
    const float* gx = Gx + (size_t)t * BB * G4H;
    v8f acc[2][4];
    // seed accumulators with precomputed x-path gates (+biases), NT reads
#pragma unroll
    for (int mt = 0; mt < 2; ++mt)
#pragma unroll
      for (int g = 0; g < 4; ++g) {
        int col = g * HH + j0 + l15;
#pragma unroll
        for (int v = 0; v < 8; ++v) {
          int row = mt * 16 + v + khalf * 8;
          acc[mt][g][v] =
              __builtin_nontemporal_load(&gx[(size_t)row * G4H + col]);
        }
      }
    // gates += h @ W_hh^T  (K=512 -> 16 chained WMMAs per tile, 8 tiles/wave)
    for (int k = 0; k < HH; k += 32) {
      v16bf a0 = load_frag16(&h_lds[(size_t)(l15) * HH + k + khalf * 8]);
      v16bf a1 = load_frag16(&h_lds[(size_t)(16 + l15) * HH + k + khalf * 8]);
#pragma unroll
      for (int g = 0; g < 4; ++g) {
        const bf16_t* pb =
            Whh + (size_t)(g * HH + j0 + l15) * HH + k + khalf * 8;
        v16bf bf = load_frag16(pb);
        acc[0][g] = wmma_bf16(a0, bf, acc[0][g]);
        acc[1][g] = wmma_bf16(a1, bf, acc[1][g]);
      }
    }
    __syncthreads();   // all reads of h_lds done before overwrite
    // pointwise LSTM update, write new h to LDS + global Hall (bf16)
#pragma unroll
    for (int mt = 0; mt < 2; ++mt)
#pragma unroll
      for (int v = 0; v < 8; ++v) {
        int row = mt * 16 + v + khalf * 8;
        int col = j0 + l15;
        float i_ = sigmoidf_(acc[mt][0][v]);
        float f_ = sigmoidf_(acc[mt][1][v]);
        float g_ = tanhf(acc[mt][2][v]);
        float o_ = sigmoidf_(acc[mt][3][v]);
        float c = f_ * creg[mt * 8 + v] + i_ * g_;
        creg[mt * 8 + v] = c;
        float h = o_ * tanhf(c);
        bf16_t hb = (bf16_t)h;
        h_lds[(size_t)row * HH + col] = hb;
        Hall[((size_t)t * BB + row) * HH + col] = hb;
      }
    __syncthreads();   // new h visible before next step's reads
  }
}

// --- K4: logits = Hall @ E^T + b_out, [4096 x 32000], K=512 ----------------
// B slab (128 cols x 32 k, 8KB) staged in LDS via async global->LDS copies,
// double-buffered on ASYNCcnt; 8 waves share it (8x cut in B cache traffic).
// fp32 outputs stored non-temporally (524MB stream; keep E in L2).
__global__ __launch_bounds__(256) void k_gemm_logits(
    const bf16_t* __restrict__ Hall, const bf16_t* __restrict__ Ebf,
    const float* __restrict__ b_out, float* __restrict__ out) {
  __shared__ bf16_t lds_b[2][128 * SPAD];   // 2 x 10 KB

  const int tid = threadIdx.x;
  const int wave = tid >> 5;
  const int lane = tid & 31;
  const int l15 = lane & 15;
  const int khalf = lane >> 4;
  const int m0 = blockIdx.y * 128 + wave * 16;
  const int n0 = blockIdx.x * 128;

  // This thread's two 16B chunks per stage: 512 chunks = 128 rows x 4 chunks.
  const int q0 = tid, q1 = 256 + tid;
  const int r0 = q0 >> 2, c0 = (q0 & 3) * 8;
  const int r1 = q1 >> 2, c1 = (q1 & 3) * 8;
  const bf16_t* gsrc0 = Ebf + (size_t)(n0 + r0) * HH + c0;
  const bf16_t* gsrc1 = Ebf + (size_t)(n0 + r1) * HH + c1;
  uint32_t ld0[2], ld1[2];
#pragma unroll
  for (int b = 0; b < 2; ++b) {
    ld0[b] = lds_offset_of(&lds_b[b][r0 * SPAD + c0]);
    ld1[b] = lds_offset_of(&lds_b[b][r1 * SPAD + c1]);
  }

  v8f acc[8];
  for (int s = 0; s < 8; ++s) acc[s] = (v8f)(0.0f);
  const bf16_t* pa_base = Hall + (size_t)(m0 + l15) * HH + khalf * 8;

  // prologue: stage k=0 into buffer 0
  async_copy_b128(ld0[0], gsrc0);
  async_copy_b128(ld1[0], gsrc1);

  for (int ks = 0; ks < 16; ++ks) {
    const int k = ks * 32;
    const int cur = ks & 1;
    if (ks + 1 < 16) {   // issue next stage, then wait for current (<=2 left)
      async_copy_b128(ld0[cur ^ 1], gsrc0 + k + 32);
      async_copy_b128(ld1[cur ^ 1], gsrc1 + k + 32);
      wait_async_le2();
    } else {
      wait_async_0();
    }
    __syncthreads();     // staged slab visible to all 8 waves

    v16bf a = load_frag16(pa_base + k);
#pragma unroll
    for (int s = 0; s < 8; ++s) {
      const bf16_t* pb = &lds_b[cur][(s * 16 + l15) * SPAD + khalf * 8];
      acc[s] = wmma_bf16(a, load_frag16(pb), acc[s]);
    }
    __syncthreads();     // done with 'cur' before it is overwritten
  }

#pragma unroll
  for (int s = 0; s < 8; ++s) {
    int col = n0 + s * 16 + l15;
    float bias = b_out[col];
#pragma unroll
    for (int v = 0; v < 8; ++v) {
      int r = m0 + v + khalf * 8;       // r = t*B + b
      int b = r & 31, t = r >> 5;
      __builtin_nontemporal_store(acc[s][v] + bias,
                                  &out[((size_t)(b * TT + t)) * VV + col]);
    }
  }
}

// ---------------------------------------------------------------------------
extern "C" void kernel_launch(void* const* d_in, const int* in_sizes, int n_in,
                              void* d_out, int out_size, void* d_ws,
                              size_t ws_size, hipStream_t stream) {
  const float* enc   = (const float*)d_in[0];   // [32, 512]
  const int*   tgt   = (const int*)d_in[1];     // [32, 128]
  const float* E     = (const float*)d_in[2];   // [32000, 512]
  const float* W_ih  = (const float*)d_in[3];   // [2048, 512]
  const float* W_hh  = (const float*)d_in[4];   // [2048, 512]
  const float* b_ih  = (const float*)d_in[5];   // [2048]
  const float* b_hh  = (const float*)d_in[6];   // [2048]
  const float* b_out = (const float*)d_in[7];   // [32000]
  float* out = (float*)d_out;                   // [32, 128, 32000]

  // workspace carve-up (256B aligned)
  auto align = [](size_t x) { return (x + 255) & ~(size_t)255; };
  char* ws = (char*)d_ws;
  size_t off = 0;
  bf16_t* Ebf  = (bf16_t*)(ws + off); off += align((size_t)VV * HH * 2);
  bf16_t* Wihb = (bf16_t*)(ws + off); off += align((size_t)G4H * HH * 2);
  bf16_t* Whhb = (bf16_t*)(ws + off); off += align((size_t)G4H * HH * 2);
  bf16_t* Xbf  = (bf16_t*)(ws + off); off += align((size_t)TT * BB * HH * 2);
  bf16_t* Hall = (bf16_t*)(ws + off); off += align((size_t)TT * BB * HH * 2);
  float*  Gx   = (float*)(ws + off);  off += align((size_t)TT * BB * G4H * 4);

  // K0: precision conversion (one pass, HBM-bound, ~66MB read)
  k_cvt_bf16<<<4096, 256, 0, stream>>>(E, Ebf, (size_t)VV * HH);
  k_cvt_bf16<<<512, 256, 0, stream>>>(W_ih, Wihb, (size_t)G4H * HH);
  k_cvt_bf16<<<512, 256, 0, stream>>>(W_hh, Whhb, (size_t)G4H * HH);
  // K1: teacher-forced input gather
  k_gather_x<<<TT * BB, 256, 0, stream>>>(E, tgt, Xbf);
  // K2: x-path gates for all timesteps (WMMA GEMM 4096x2048, K=512)
  k_gemm_gates<<<dim3(G4H / 128, (TT * BB) / 128), 256, 0, stream>>>(
      Xbf, Wihb, b_ih, b_hh, Gx);
  // K3: sequential recurrence, single WGP persistent workgroup
  k_recurrence<<<1, 1024, 0, stream>>>(enc, Gx, Whhb, Hall);
  // K4: tied-embedding projection (WMMA GEMM 4096x32000, K=512) + permute
  k_gemm_logits<<<dim3(VV / 128, (TT * BB) / 128), 256, 0, stream>>>(
      Hall, Ebf, b_out, out);
}
